// TLSTM_22522808500677
// MI455X (gfx1250) — matrix operand order repaired and checked by reading, not compiled, not run on hardware
//
#include <hip/hip_runtime.h>
#include <math.h>

// ---------------------------------------------------------------------------
// TLSTM for MI455X (gfx1250, wave32, WMMA bf16 16x16x32).
//   B=256, T=512, I=512, H=256
// K1: pack fp32 weights -> bf16 WMMA B-fragments.
// K2: input projections; 32-row blocks; depth-4 B-fragment prefetch ring.
// K3: sequential scan; persistent depth-10 B-fragment ring across kt AND t;
//     double-buffered async X staging; fused gate math.
// ---------------------------------------------------------------------------

#define BB 256
#define TT 512
#define II 512
#define HH 256

typedef __attribute__((ext_vector_type(16))) __bf16       v16bf;
typedef __attribute__((ext_vector_type(8)))  float        v8f;
typedef __attribute__((ext_vector_type(4)))  unsigned int u32x4;
typedef __attribute__((ext_vector_type(4)))  float        f32x4;

#define AS1 __attribute__((address_space(1)))
#define AS3 __attribute__((address_space(3)))

#if defined(__HIP_DEVICE_COMPILE__) && \
    __has_builtin(__builtin_amdgcn_global_load_async_to_lds_b128) && \
    __has_builtin(__builtin_amdgcn_s_wait_asynccnt)
#define USE_ASYNC 1
#else
#define USE_ASYNC 0
#endif

#if USE_ASYNC
typedef int v4i_gcc __attribute__((vector_size(16)));
__device__ __forceinline__ void async_b128(const void* g, void* l) {
    __builtin_amdgcn_global_load_async_to_lds_b128(
        (AS1 v4i_gcc*)(AS1 void*)(void*)g,
        (AS3 v4i_gcc*)(AS3 void*)l, 0, 0);
}
#endif

union Frag {
    v16bf          v;
    u32x4          u[2];
    unsigned short s[16];
};

__device__ __forceinline__ unsigned short f2bf(float f) {
    unsigned int u = __builtin_bit_cast(unsigned int, f);
    u += 0x7FFFu + ((u >> 16) & 1u);   // round-to-nearest-even
    return (unsigned short)(u >> 16);
}
__device__ __forceinline__ float bf2f(unsigned short s) {
    unsigned int u = ((unsigned int)s) << 16;
    return __builtin_bit_cast(float, u);
}
__device__ __forceinline__ float sigmf(float x) {
    return 1.0f / (1.0f + __expf(-x));
}
__device__ __forceinline__ v8f wmma_bf16(v16bf a, v16bf b, v8f c) {
    return __builtin_amdgcn_wmma_f32_16x16x32_bf16(
        false, a, false, b, (short)0, c, false, false);
}
// Build a bf16 A-fragment (16x32) from an fp32 LDS row.
__device__ __forceinline__ void build_afrag_f32(Frag& a, const float* row, int kbase) {
    #pragma unroll
    for (int g = 0; g < 2; ++g) {
        f32x4 f0 = *(const f32x4*)(row + kbase + g * 16);
        f32x4 f1 = *(const f32x4*)(row + kbase + g * 16 + 4);
        a.s[g * 8 + 0] = f2bf(f0.x); a.s[g * 8 + 1] = f2bf(f0.y);
        a.s[g * 8 + 2] = f2bf(f0.z); a.s[g * 8 + 3] = f2bf(f0.w);
        a.s[g * 8 + 4] = f2bf(f1.x); a.s[g * 8 + 5] = f2bf(f1.y);
        a.s[g * 8 + 6] = f2bf(f1.z); a.s[g * 8 + 7] = f2bf(f1.w);
    }
}

// Workspace layout (ushort units):
#define WXF_OFF  0           // 4*16*16*512          = 524288
#define WMF_OFF  524288      // 5*8*16*512           = 327680
#define XBUF_OFF 851968      // B*T*4*H bf16         = 134217728

// ---------------------------------------------------------------------------
// Kernel 1: pack weights into bf16 B-fragment layout.
// ---------------------------------------------------------------------------
__global__ void __launch_bounds__(32)
pack_weights(const float* wix, const float* wfx, const float* wcx, const float* wox,
             const float* wim, const float* wfm, const float* wcm, const float* wom,
             const float* wtm, unsigned short* ws16) {
    const int lane = threadIdx.x & 31;
    const int n = lane & 15, halfL = lane >> 4;
    const float* wx[4] = {wix, wfx, wcx, wox};
    const float* wm[5] = {wim, wfm, wcm, wom, wtm};

    int bi = blockIdx.x;
    if (bi < 1024) {
        int gate = bi >> 8, rem = bi & 255, kt = rem >> 4, nt = rem & 15;
        const float* W = wx[gate];
        unsigned short* dst =
            ws16 + WXF_OFF + (((size_t)(gate * 16 + kt) * 16 + nt) * 512) + lane * 16;
        #pragma unroll
        for (int j = 0; j < 8; ++j)
            #pragma unroll
            for (int e = 0; e < 2; ++e) {
                int k = kt * 32 + (j >> 2) * 16 + halfL * 8 + (j & 3) * 2 + e;
                dst[j * 2 + e] = f2bf(W[(size_t)(nt * 16 + n) * II + k]);
            }
    } else {
        int bi2 = bi - 1024;
        int mat = bi2 >> 7, rem = bi2 & 127, kt = rem >> 4, nt = rem & 15;
        const float* W = wm[mat];
        unsigned short* dst =
            ws16 + WMF_OFF + (((size_t)(mat * 8 + kt) * 16 + nt) * 512) + lane * 16;
        #pragma unroll
        for (int j = 0; j < 8; ++j)
            #pragma unroll
            for (int e = 0; e < 2; ++e) {
                int k = kt * 32 + (j >> 2) * 16 + halfL * 8 + (j & 3) * 2 + e;
                dst[j * 2 + e] = f2bf(W[(size_t)(nt * 16 + n) * HH + k]);
            }
    }
}

// ---------------------------------------------------------------------------
// Kernel 2: input projections. Block = 32 rows (two 16-row M tiles); 8 waves,
// each wave owns 8 (gate,ntile) tiles x 2 M tiles. B-fragments flow through a
// depth-4 prefetch ring (consumed 4 steps / 8 WMMAs after issue) so L2
// latency is hidden and ring slots cannot be register-coalesced.
// Dynamic LDS: 32 * 520 floats = 66560 B.
// ---------------------------------------------------------------------------
__global__ void __launch_bounds__(256, 1)
input_proj(const float* __restrict__ feat,
           const float* __restrict__ bix, const float* __restrict__ bfx,
           const float* __restrict__ bcx, const float* __restrict__ box,
           const unsigned short* __restrict__ ws16) {
    extern __shared__ char smem[];
    float* a_lds = (float*)smem;                    // [32][520] padded fp32

    const int tid  = threadIdx.x;
    const int lane = tid & 31;
    const int wave = tid >> 5;
    const int n    = lane & 15;
    const int halfL = lane >> 4;

    const unsigned short* wxf = ws16 + WXF_OFF;
    unsigned short* xbuf = const_cast<unsigned short*>(ws16) + XBUF_OFF;

    const float* fbase = feat + (size_t)blockIdx.x * 32 * II;
#if USE_ASYNC
    for (int idx = tid; idx < 32 * 128; idx += 256) {
        int r = idx >> 7, c4 = idx & 127;
        async_b128(fbase + (size_t)r * II + c4 * 4,
                   a_lds + r * 520 + c4 * 4);
    }
    __builtin_amdgcn_s_wait_asynccnt(0);
#else
    for (int idx = tid; idx < 32 * 128; idx += 256) {
        int r = idx >> 7, c4 = idx & 127;
        *(f32x4*)(a_lds + r * 520 + c4 * 4) =
            *(const f32x4*)(fbase + (size_t)r * II + c4 * 4);
    }
#endif
    __syncthreads();

    v8f acc[2][8];
    #pragma unroll
    for (int r = 0; r < 2; ++r)
        #pragma unroll
        for (int p = 0; p < 8; ++p) acc[r][p] = (v8f){0,0,0,0,0,0,0,0};

    const float* arow0 = a_lds + (0 * 16 + n) * 520;
    const float* arow1 = a_lds + (1 * 16 + n) * 520;

    // B-fragment loader: flattened step idx = kt*8 + p
    Frag bring[4];
    auto load_bx = [&](int kt2, int p2, Frag& dst) {
        int tile = wave * 8 + p2;
        int gate = tile >> 4, nt = tile & 15;
        const u32x4* bp = (const u32x4*)&wxf[((size_t)(gate * 16 + kt2) * 16 + nt) * 512];
        dst.u[0] = bp[lane * 2];
        dst.u[1] = bp[lane * 2 + 1];
    };
    // prologue: fill ring with steps 0..3 (kt=0, p=0..3)
    #pragma unroll
    for (int s = 0; s < 4; ++s) load_bx(0, s, bring[s]);

    for (int kt = 0; kt < 16; ++kt) {
        Frag a0, a1;
        build_afrag_f32(a0, arow0, kt * 32 + halfL * 8);
        build_afrag_f32(a1, arow1, kt * 32 + halfL * 8);
        #pragma unroll
        for (int p = 0; p < 8; ++p) {
            Frag& bc = bring[p & 3];
            acc[0][p] = wmma_bf16(a0.v, bc.v, acc[0][p]);
            acc[1][p] = wmma_bf16(a1.v, bc.v, acc[1][p]);
            // prefetch step idx+4 into the slot just consumed (same slot id)
            int pn  = (p + 4) & 7;
            int ktn = (kt + ((p + 4) >> 3)) & 15;   // wrap at end: dead loads
            load_bx(ktn, pn, bc);
        }
    }

    const float* bias[4] = {bix, bfx, bcx, box};
    #pragma unroll
    for (int r = 0; r < 2; ++r)
        #pragma unroll
        for (int p = 0; p < 8; ++p) {
            int tile = wave * 8 + p;
            int gate = tile >> 4, nt = tile & 15;
            int col  = nt * 16 + n;
            float bb = bias[gate][col];
            #pragma unroll
            for (int j = 0; j < 8; ++j) {
                int m = halfL * 8 + j;
                size_t bt = (size_t)blockIdx.x * 32 + r * 16 + m;
                xbuf[(bt * 4 + gate) * HH + col] = f2bf(acc[r][p][j] + bb);
            }
        }
}

// ---------------------------------------------------------------------------
// Kernel 3: recurrent scan. 16 blocks x 256 threads; block owns 16 batch rows.
// Dynamic LDS layout (95808 B):
//   [0)      x double buffer 2*16*1024 bf16 = 65536
//   [65536)  h bf16 [16][264]               =  8448
//   [73984)  c fp32 [16][260]               = 16640
//   [90624)  biases 5*256 fp32              =  5120
//   [95744)  disc 16 fp32                   =    64
// B-fragments live in a persistent depth-10 ring: WMMA (kt,u) consumes slot u
// then immediately prefetches (kt+1,u) -> 10-WMMA prefetch distance; the kt=7
// prefetch wraps to kt=0, keeping the ring warm for the next time step.
// ---------------------------------------------------------------------------
__global__ void __launch_bounds__(256, 1)
scan(const float* __restrict__ timep,
     const unsigned short* __restrict__ ws16,
     const float* __restrict__ bim, const float* __restrict__ bfm,
     const float* __restrict__ bcm, const float* __restrict__ bom,
     const float* __restrict__ btm,
     float* __restrict__ out) {
    extern __shared__ char smem[];
    unsigned short* x_buf[2];
    x_buf[0] = (unsigned short*)smem;
    x_buf[1] = (unsigned short*)smem + 16 * 1024;
    unsigned short* h_lds    = (unsigned short*)(smem + 65536);  // stride 264
    float*          c_lds    = (float*)(smem + 73984);           // stride 260
    float*          bias_lds = (float*)(smem + 90624);
    float*          disc_lds = (float*)(smem + 95744);

    const int tid  = threadIdx.x;
    const int lane = tid & 31;
    const int wave = tid >> 5;
    const int n    = lane & 15;
    const int halfL = lane >> 4;
    const int b0   = blockIdx.x * 16;
    const int nt0  = wave * 2;

    const unsigned short* wmf  = ws16 + WMF_OFF;
    const unsigned short* xbuf = ws16 + XBUF_OFF;

    // ---- prologue: kick off X(0) staging, then init state/biases
    {
#if USE_ASYNC
        for (int idx = tid; idx < 16 * 128; idx += 256) {
            int m = idx >> 7, c8 = idx & 127;
            async_b128(&xbuf[(((size_t)(b0 + m) * TT + 0) * 4) * HH + c8 * 8],
                       &x_buf[0][m * 1024 + c8 * 8]);
        }
#else
        for (int idx = tid; idx < 16 * 128; idx += 256) {
            int m = idx >> 7, c8 = idx & 127;
            *(u32x4*)&x_buf[0][m * 1024 + c8 * 8] =
                *(const u32x4*)&xbuf[(((size_t)(b0 + m) * TT + 0) * 4) * HH + c8 * 8];
        }
#endif
        for (int i = tid; i < 16 * 264; i += 256) h_lds[i] = 0;
        for (int i = tid; i < 16 * 260; i += 256) c_lds[i] = 0.0f;
        const float* bptr[5] = {bim, bfm, bcm, bom, btm};
        for (int i = tid; i < 5 * HH; i += 256) bias_lds[i] = bptr[i >> 8][i & 255];
    }

    // ---- persistent B-fragment ring (slot u = mat*2+q), preload kt=0
    Frag bring[10];
    auto load_bm = [&](int kt2, int u2, Frag& dst) {
        int mat = u2 >> 1, q = u2 & 1;
        const u32x4* bp = (const u32x4*)
            &wmf[((size_t)(mat * 8 + kt2) * 16 + (nt0 + q)) * 512];
        dst.u[0] = bp[lane * 2];
        dst.u[1] = bp[lane * 2 + 1];
    };
    #pragma unroll
    for (int u = 0; u < 10; ++u) load_bm(0, u, bring[u]);

    for (int t = 0; t < TT; ++t) {
        const unsigned short* xc = x_buf[t & 1];
        unsigned short*       xn = x_buf[(t & 1) ^ 1];

        // ---- stage next step's X block into the other buffer
        if (t + 1 < TT) {
#if USE_ASYNC
            for (int idx = tid; idx < 16 * 128; idx += 256) {
                int m = idx >> 7, c8 = idx & 127;
                async_b128(&xbuf[(((size_t)(b0 + m) * TT + (t + 1)) * 4) * HH + c8 * 8],
                           &xn[m * 1024 + c8 * 8]);
            }
#else
            for (int idx = tid; idx < 16 * 128; idx += 256) {
                int m = idx >> 7, c8 = idx & 127;
                *(u32x4*)&xn[m * 1024 + c8 * 8] =
                    *(const u32x4*)&xbuf[(((size_t)(b0 + m) * TT + (t + 1)) * 4) * HH + c8 * 8];
            }
#endif
        }
        if (tid < 16) {
            float tv = timep[(size_t)(b0 + tid) * TT + t];
            disc_lds[tid] = 1.0f / logf(2.718281828459045f + tv);
        }
#if USE_ASYNC
        if (t + 1 < TT) __builtin_amdgcn_s_wait_asynccnt(8);
        else            __builtin_amdgcn_s_wait_asynccnt(0);
#endif
        __syncthreads();   // (A) X(t), disc visible; prev h/c writes ordered

        // ---- 5 recurrent GEMMs with rolling B prefetch
        v8f acc[2][5];
        #pragma unroll
        for (int q = 0; q < 2; ++q)
            #pragma unroll
            for (int mmat = 0; mmat < 5; ++mmat) acc[q][mmat] = (v8f){0,0,0,0,0,0,0,0};

        for (int kt = 0; kt < 8; ++kt) {
            Frag ah, ac;
            ah.u[0] = *(const u32x4*)&h_lds[n * 264 + kt * 32 + halfL * 8];
            ah.u[1] = *(const u32x4*)&h_lds[n * 264 + kt * 32 + 16 + halfL * 8];
            build_afrag_f32(ac, c_lds + n * 260, kt * 32 + halfL * 8);

            #pragma unroll
            for (int u = 0; u < 10; ++u) {
                int mmat = u >> 1, q = u & 1;
                acc[q][mmat] = wmma_bf16(mmat == 4 ? ac.v : ah.v,
                                         bring[u].v, acc[q][mmat]);
                load_bm((kt + 1) & 7, u, bring[u]);   // wraps to kt=0 at end
            }
        }

        // ---- fused gate math into registers
        float hy_r[2][8], cy_r[2][8];
        #pragma unroll
        for (int q = 0; q < 2; ++q) {
            int col = (nt0 + q) * 16 + n;
            #pragma unroll
            for (int j = 0; j < 8; ++j) {
                int m = halfL * 8 + j;
                float c_old = c_lds[m * 260 + col];
                float d     = disc_lds[m];
                float cs    = tanhf(acc[q][4][j] + bias_lds[4 * HH + col]);
                float c_adj = (c_old - cs) + d * cs;
                float xi = bf2f(xc[m * 1024 + 0 * HH + col]);
                float xf = bf2f(xc[m * 1024 + 1 * HH + col]);
                float xg = bf2f(xc[m * 1024 + 2 * HH + col]);
                float xo = bf2f(xc[m * 1024 + 3 * HH + col]);
                float iv = sigmf(xi + acc[q][0][j] + bias_lds[0 * HH + col]);
                float fv = sigmf(xf + acc[q][1][j] + bias_lds[1 * HH + col]);
                float gv = tanhf(xg + acc[q][2][j] + bias_lds[2 * HH + col]);
                float ov = sigmf(xo + acc[q][3][j] + bias_lds[3 * HH + col]);
                float cy = fv * c_adj + iv * gv;
                float hy = ov * tanhf(cy);
                cy_r[q][j] = cy;
                hy_r[q][j] = hy;
            }
        }
        __syncthreads();   // (B) all waves done reading old h/c/x

        #pragma unroll
        for (int q = 0; q < 2; ++q) {
            int col = (nt0 + q) * 16 + n;
            #pragma unroll
            for (int j = 0; j < 8; ++j) {
                int m = halfL * 8 + j;
                h_lds[m * 264 + col] = f2bf(hy_r[q][j]);
                c_lds[m * 260 + col] = cy_r[q][j];
                out[((size_t)(b0 + m) * TT + t) * HH + col] = hy_r[q][j];
                if (t == TT - 1)
                    out[(size_t)BB * TT * HH + (size_t)(b0 + m) * HH + col] = hy_r[q][j];
            }
        }
        __syncthreads();   // (C) h/c visible before next step
    }
}

// ---------------------------------------------------------------------------
extern "C" void kernel_launch(void* const* d_in, const int* in_sizes, int n_in,
                              void* d_out, int out_size, void* d_ws, size_t ws_size,
                              hipStream_t stream) {
    (void)in_sizes; (void)n_in; (void)out_size; (void)ws_size;
    const float* feature = (const float*)d_in[0];
    const float* timep   = (const float*)d_in[1];
    const float* W_ix = (const float*)d_in[2];  const float* b_ix = (const float*)d_in[3];
    const float* W_fx = (const float*)d_in[4];  const float* b_fx = (const float*)d_in[5];
    const float* W_cx = (const float*)d_in[6];  const float* b_cx = (const float*)d_in[7];
    const float* W_ox = (const float*)d_in[8];  const float* b_ox = (const float*)d_in[9];
    const float* W_im = (const float*)d_in[10]; const float* b_im = (const float*)d_in[11];
    const float* W_fm = (const float*)d_in[12]; const float* b_fm = (const float*)d_in[13];
    const float* W_cm = (const float*)d_in[14]; const float* b_cm = (const float*)d_in[15];
    const float* W_om = (const float*)d_in[16]; const float* b_om = (const float*)d_in[17];
    const float* W_tm = (const float*)d_in[18]; const float* b_tm = (const float*)d_in[19];

    unsigned short* ws16 = (unsigned short*)d_ws;

    pack_weights<<<1664, 32, 0, stream>>>(W_ix, W_fx, W_cx, W_ox,
                                          W_im, W_fm, W_cm, W_om, W_tm, ws16);

    input_proj<<<(BB * TT) / 32, 256, 32 * 520 * 4, stream>>>(
        feature, b_ix, b_fx, b_cx, b_ox, ws16);

    scan<<<BB / 16, 256, 95808, stream>>>(timep, ws16, b_im, b_fm, b_cm, b_om, b_tm,
                                          (float*)d_out);
}
